// MusicTransformerContinuousToken_71339406786619
// MI455X (gfx1250) — matrix operand order repaired and checked by reading, not compiled, not run on hardware
//
#include <hip/hip_runtime.h>
#include <math.h>
#include <stdint.h>

// ---------------- problem constants ----------------
#define B_      2
#define S_TOK_  2046
#define NC_     2
#define S_      2048
#define D_      512
#define DI_     2048
#define V_      512
#define L_      6
#define H_      8
#define DH_     64
#define MS_     (B_ * S_)
#define NEG_BIG (-3.0e38f)

typedef __bf16 bf16_t;
typedef bf16_t v16bf __attribute__((ext_vector_type(16)));
typedef bf16_t v8bf  __attribute__((ext_vector_type(8)));
typedef bf16_t v4bf  __attribute__((ext_vector_type(4)));
typedef float  v8f   __attribute__((ext_vector_type(8)));

// A-fragment K pattern (16-bit A 16x32): lanes 0-15 hold K0-15, lanes 16-31 hold K16-31 (order used here, same as B).
// B-fragment K pattern (16-bit B 32x16): lanes 0-15 hold K0-15, lanes 16-31 hold K16-31.
__device__ __forceinline__ v16bf cat8(v8bf a, v8bf b) {
    return __builtin_shufflevector(a, b, 0,1,2,3,4,5,6,7,8,9,10,11,12,13,14,15);
}

// gfx1250 async DMA: copy 16B/lane from global to LDS, tracked by ASYNCcnt.
__device__ __forceinline__ void async_copy_b128(void* lds_dst, const void* gsrc) {
    unsigned ldsa = (unsigned)(uintptr_t)lds_dst;   // LDS aperture: low 32 bits = LDS offset
    asm volatile("global_load_async_to_lds_b128 %0, %1, off"
                 :: "v"(ldsa), "v"(gsrc) : "memory");
}
__device__ __forceinline__ void async_wait0() {
    asm volatile("s_wait_asynccnt 0" ::: "memory");
}

// ---------------- generic tiled WMMA GEMM (bf16 operands, fp32 accum) ----------------
// mode 0: Cf fp32 [M,N];  mode 1: Cb bf16 [M,N];  mode 2: Cb bf16 V-transposed [b][h][dh][s]
#define TM 64
#define TN 64
#define TK 32

__global__ __launch_bounds__(256)
void gemm_bf16_wmma(const bf16_t* __restrict__ A, const bf16_t* __restrict__ Bw,
                    const float* __restrict__ bias, const float* __restrict__ addmat,
                    float* __restrict__ Cf, bf16_t* __restrict__ Cb,
                    int M, int N, int K, int relu, int mode)
{
    __shared__ bf16_t As[TM][TK + 8];
    __shared__ bf16_t BsT[TN][TK + 8];   // transposed: fragment K-runs contiguous

    const int tid  = threadIdx.x;
    const int lane = tid & 31;
    const int wave = tid >> 5;
    const int half = lane >> 4;
    const int ln   = lane & 15;

    const int m0 = blockIdx.y * TM;
    const int n0 = blockIdx.x * TN;
    const int fm = (wave & 3) * 16;
    const int fn = (wave >> 2) * 32;

    const int ar_  = tid >> 2, aseg = tid & 3;   // A stage: 64 rows x 4 16B-chunks
    const int br_  = tid >> 3, bseg = tid & 7;   // B stage: 32 rows x 8 8-elem chunks

    v8f acc0 = {}; v8f acc1 = {};

    for (int k0 = 0; k0 < K; k0 += TK) {
        // A tile: raw bf16 -> async DMA to LDS (one b128 per lane)
        async_copy_b128(&As[ar_][aseg * 8],
                        A + (size_t)(m0 + ar_) * K + k0 + aseg * 8);
        // B tile: vector load + transpose scatter into BsT
        v8bf bv = *(const v8bf*)(Bw + (size_t)(k0 + br_) * N + n0 + bseg * 8);
#pragma unroll
        for (int i2 = 0; i2 < 8; ++i2) BsT[bseg * 8 + i2][br_] = bv[i2];
        async_wait0();
        __syncthreads();

        const bf16_t* arp = &As[fm + ln][0];
        v16bf af = cat8(*(const v8bf*)(arp + half * 16),
                        *(const v8bf*)(arp + half * 16 + 8));
        const bf16_t* br0 = &BsT[fn + ln][half * 16];
        v16bf bf0 = cat8(*(const v8bf*)br0, *(const v8bf*)(br0 + 8));
        const bf16_t* br1 = &BsT[fn + 16 + ln][half * 16];
        v16bf bf1 = cat8(*(const v8bf*)br1, *(const v8bf*)(br1 + 8));

        acc0 = __builtin_amdgcn_wmma_f32_16x16x32_bf16(false, af, false, bf0,
                                                       (short)0, acc0, false, false);
        acc1 = __builtin_amdgcn_wmma_f32_16x16x32_bf16(false, af, false, bf1,
                                                       (short)0, acc1, false, false);
        __syncthreads();
    }

#pragma unroll
    for (int r = 0; r < 8; ++r) {
        int row  = m0 + fm + r + half * 8;
        int col0 = n0 + fn + ln;
        int col1 = col0 + 16;
        float v0 = acc0[r] + bias[col0];
        float v1 = acc1[r] + bias[col1];
        if (relu) { v0 = fmaxf(v0, 0.f); v1 = fmaxf(v1, 0.f); }
        if (addmat) {
            v0 += addmat[(size_t)row * N + col0];
            v1 += addmat[(size_t)row * N + col1];
        }
        if (mode == 0) {
            Cf[(size_t)row * N + col0] = v0;
            Cf[(size_t)row * N + col1] = v1;
        } else if (mode == 1) {
            Cb[(size_t)row * N + col0] = (bf16_t)v0;
            Cb[(size_t)row * N + col1] = (bf16_t)v1;
        } else {  // V-transposed: [b][h][dh][s]
            int b2 = row >> 11, s = row & (S_ - 1);
            int h0 = col0 >> 6, d0 = col0 & 63;
            int h1 = col1 >> 6, d1 = col1 & 63;
            Cb[(((size_t)(b2 * H_ + h0) * DH_ + d0) << 11) + s] = (bf16_t)v0;
            Cb[(((size_t)(b2 * H_ + h1) * DH_ + d1) << 11) + s] = (bf16_t)v1;
        }
    }
}

// ---------------- flash attention with relative positions (all-bf16 operands) ----------------
__device__ __forceinline__ float key_bias(const int* toks, int b, int j) {
    if (j < NC_) return 0.f;
    return (toks[b * S_TOK_ + (j - NC_)] == 0) ? NEG_BIG : 0.f;
}

__global__ __launch_bounds__(32)
void attn_flash_rel(const bf16_t* __restrict__ Q, const bf16_t* __restrict__ Kb,
                    const bf16_t* __restrict__ Vt, const bf16_t* __restrict__ E,
                    const int* __restrict__ toks, bf16_t* __restrict__ O)
{
    __shared__ bf16_t Pl[16][40];

    const int lane = threadIdx.x;
    const int half = lane >> 4;
    const int ln   = lane & 15;

    const int qb = blockIdx.x % (S_ / 16);
    const int h  = (blockIdx.x / (S_ / 16)) % H_;
    const int b  = blockIdx.x / ((S_ / 16) * H_);
    const int i0 = qb * 16;

    const bf16_t* qbase = Q  + (size_t)(b * S_) * D_ + h * DH_;
    const bf16_t* kbase = Kb + (size_t)(b * S_) * D_ + h * DH_;
    const bf16_t* vbase = Vt + (size_t)(b * H_ + h) * DH_ * S_;   // [dh][s]

    // Q fragments (A-layout), DH=64 -> two K-steps
    const bf16_t* qr = qbase + (size_t)(i0 + ln) * D_;
    v16bf q0 = cat8(*(const v8bf*)(qr + half * 16),      *(const v8bf*)(qr + 8 + half * 16));
    v16bf q1 = cat8(*(const v8bf*)(qr + 32 + half * 16), *(const v8bf*)(qr + 40 + half * 16));

    v8f o0 = {}, o1 = {}, o2 = {}, o3 = {};
    float mrun[8], lrun[8];
#pragma unroll
    for (int r = 0; r < 8; ++r) { mrun[r] = NEG_BIG; lrun[r] = 0.f; }

    for (int j0 = 0; j0 < i0 + 16; j0 += 32) {   // keys j0..j0+31, always <= S-1
        // ---- logits tiles: S = Q K^T ----
        v8f s0 = {}, s1 = {};
#pragma unroll
        for (int t = 0; t < 2; ++t) {
            v8f st = {};
#pragma unroll
            for (int ks = 0; ks < 2; ++ks) {
                const bf16_t* kr = kbase + (size_t)(j0 + t * 16 + ln) * D_ + ks * 32 + half * 16;
                v16bf kf = cat8(*(const v8bf*)kr, *(const v8bf*)(kr + 8));
                st = __builtin_amdgcn_wmma_f32_16x16x32_bf16(false, ks ? q1 : q0, false, kf,
                                                             (short)0, st, false, false);
            }
            if (t == 0) s0 = st; else s1 = st;
        }

        // ---- G = Q @ E_band^T over 47-row diagonal band ----
        const int ebase = S_ - 1 - i0 + j0 - 15;
        v8f G0 = {}, G1 = {}, G2 = {};
#pragma unroll
        for (int g = 0; g < 3; ++g) {
            int eidx = ebase + g * 16 + ln;
            if (eidx > S_ - 1) eidx = S_ - 1;    // masked region only
            v8f gt = {};
#pragma unroll
            for (int ks = 0; ks < 2; ++ks) {
                const bf16_t* ep = E + (size_t)eidx * DH_ + ks * 32 + half * 16;
                v16bf ef = cat8(*(const v8bf*)ep, *(const v8bf*)(ep + 8));
                gt = __builtin_amdgcn_wmma_f32_16x16x32_bf16(false, ks ? q1 : q0, false, ef,
                                                             (short)0, gt, false, false);
            }
            if (g == 0) G0 = gt; else if (g == 1) G1 = gt; else G2 = gt;
        }

        // ---- merge Srel (lane-permute from C-layout), bias, causal ----
        float kb0 = key_bias(toks, b, j0 + ln);
        float kb1 = key_bias(toks, b, j0 + 16 + ln);
        float l0[8], l1[8];
#pragma unroll
        for (int r = 0; r < 8; ++r) {
            const int M = r + 8 * half;
            const int i = i0 + M;
            {
                int c = 15 + ln - M;                 // [0,30]
                int srcl = half * 16 + (c & 15);
                float ga = __shfl(G0[r], srcl, 32);
                float gb = __shfl(G1[r], srcl, 32);
                float lv = (s0[r] + ((c < 16) ? ga : gb)) * 0.125f + kb0;
                if (j0 + ln > i) lv = NEG_BIG;
                l0[r] = lv;
            }
            {
                int c = 31 + ln - M;                 // [16,46]
                int srcl = half * 16 + (c & 15);
                float ga = __shfl(G1[r], srcl, 32);
                float gb = __shfl(G2[r], srcl, 32);
                float lv = (s1[r] + ((c < 32) ? ga : gb)) * 0.125f + kb1;
                if (j0 + 16 + ln > i) lv = NEG_BIG;
                l1[r] = lv;
            }
        }

        // ---- online softmax ----
#pragma unroll
        for (int r = 0; r < 8; ++r) {
            float vm = fmaxf(l0[r], l1[r]);
#pragma unroll
            for (int m = 8; m >= 1; m >>= 1) vm = fmaxf(vm, __shfl_xor(vm, m, 32));
            float mnew = fmaxf(mrun[r], vm);
            float p0 = (l0[r] <= -1.0e37f) ? 0.f : expf(l0[r] - mnew);
            float p1 = (l1[r] <= -1.0e37f) ? 0.f : expf(l1[r] - mnew);
            float ps = p0 + p1;
#pragma unroll
            for (int m = 8; m >= 1; m >>= 1) ps += __shfl_xor(ps, m, 32);
            float scale = expf(mrun[r] - mnew);
            lrun[r] = lrun[r] * scale + ps;
            mrun[r] = mnew;
            o0[r] *= scale; o1[r] *= scale; o2[r] *= scale; o3[r] *= scale;
            Pl[r + 8 * half][ln]      = (bf16_t)p0;
            Pl[r + 8 * half][16 + ln] = (bf16_t)p1;
        }
        __syncthreads();

        // ---- O += P V : V fragments contiguous thanks to [dh][s] layout ----
        const bf16_t* prp = &Pl[ln][0];
        v16bf pf = cat8(*(const v8bf*)(prp + half * 16), *(const v8bf*)(prp + half * 16 + 8));
#pragma unroll
        for (int t = 0; t < 4; ++t) {
            const bf16_t* vr = vbase + (size_t)(t * 16 + ln) * S_ + j0 + half * 16;
            v16bf vf = cat8(*(const v8bf*)vr, *(const v8bf*)(vr + 8));
            v8f* op = (t == 0) ? &o0 : (t == 1) ? &o1 : (t == 2) ? &o2 : &o3;
            *op = __builtin_amdgcn_wmma_f32_16x16x32_bf16(false, pf, false, vf,
                                                          (short)0, *op, false, false);
        }
        __syncthreads();
    }

#pragma unroll
    for (int r = 0; r < 8; ++r) {
        int row = i0 + r + 8 * half;
        float inv = 1.0f / lrun[r];
        bf16_t* op = O + (size_t)(b * S_ + row) * D_ + h * DH_;
        op[ln]      = (bf16_t)(o0[r] * inv);
        op[16 + ln] = (bf16_t)(o1[r] * inv);
        op[32 + ln] = (bf16_t)(o2[r] * inv);
        op[48 + ln] = (bf16_t)(o3[r] * inv);
    }
}

// ---------------- layernorm: fp32 in, fp32 + optional bf16 out ----------------
__global__ __launch_bounds__(256)
void layernorm_k(const float* __restrict__ X, const float* __restrict__ g,
                 const float* __restrict__ bta, float* __restrict__ Y,
                 bf16_t* __restrict__ Yb)
{
    __shared__ float red[8];
    const int row = blockIdx.x;
    const int tid = threadIdx.x;
    const float* xr = X + (size_t)row * D_;
    float a = xr[tid], c = xr[tid + 256];
    float s = a + c;
#pragma unroll
    for (int m = 16; m >= 1; m >>= 1) s += __shfl_xor(s, m, 32);
    if ((tid & 31) == 0) red[tid >> 5] = s;
    __syncthreads();
    float tot = 0.f;
#pragma unroll
    for (int i = 0; i < 8; ++i) tot += red[i];
    float mu = tot * (1.0f / D_);
    __syncthreads();
    float d0 = a - mu, d1 = c - mu;
    float sv = d0 * d0 + d1 * d1;
#pragma unroll
    for (int m = 16; m >= 1; m >>= 1) sv += __shfl_xor(sv, m, 32);
    if ((tid & 31) == 0) red[tid >> 5] = sv;
    __syncthreads();
    float tv = 0.f;
#pragma unroll
    for (int i = 0; i < 8; ++i) tv += red[i];
    float rstd = rsqrtf(tv * (1.0f / D_) + 1e-6f);
    float y0 = d0 * rstd * g[tid] + bta[tid];
    float y1 = d1 * rstd * g[tid + 256] + bta[tid + 256];
    Y[(size_t)row * D_ + tid]       = y0;
    Y[(size_t)row * D_ + tid + 256] = y1;
    if (Yb) {
        Yb[(size_t)row * D_ + tid]       = (bf16_t)y0;
        Yb[(size_t)row * D_ + tid + 256] = (bf16_t)y1;
    }
}

// ---------------- fp32 -> bf16 conversion ----------------
__global__ void cvt_bf16_k(const float* __restrict__ src, bf16_t* __restrict__ dst, long n)
{
    long i = ((long)blockIdx.x * 256 + threadIdx.x) * 4;
    if (i >= n) return;
    float4 v = *(const float4*)(src + i);
    v4bf o = { (bf16_t)v.x, (bf16_t)v.y, (bf16_t)v.z, (bf16_t)v.w };
    *(v4bf*)(dst + i) = o;
}

// ---------------- condition MLP -> X[b, 0:NC, :] ----------------
__global__ __launch_bounds__(256)
void cond_k(const float* __restrict__ cond, const float* __restrict__ W1,
            const float* __restrict__ b1, const float* __restrict__ W2,
            const float* __restrict__ b2, const float* __restrict__ nullc,
            float* __restrict__ X)
{
    __shared__ float h1[256];
    const int c = blockIdx.x % NC_;
    const int b = blockIdx.x / NC_;
    const int tid = threadIdx.x;
    float cv = cond[b * NC_ + c];
    bool isn = (cv != cv);
    float cm = isn ? 0.f : cv;
    h1[tid] = fmaxf(cm * W1[c * 256 + tid] + b1[c * 256 + tid], 0.f);
    __syncthreads();
#pragma unroll
    for (int t = 0; t < 2; ++t) {
        int o2 = tid + t * 256;
        float acc = b2[c * D_ + o2];
        for (int o = 0; o < 256; ++o) acc += h1[o] * W2[(size_t)(c * 256 + o) * D_ + o2];
        if (isn) acc = nullc[c * D_ + o2];
        X[(size_t)(b * S_ + c) * D_ + o2] = acc;
    }
}

// ---------------- embedding * sqrt(D) + positional ----------------
__global__ void embed_k(const int* __restrict__ toks, const float* __restrict__ emb,
                        const float* __restrict__ pos, float* __restrict__ X)
{
    size_t idx = (size_t)blockIdx.x * 256 + threadIdx.x;
    if (idx >= (size_t)B_ * S_ * D_) return;
    int d = (int)(idx % D_);
    int s = (int)((idx / D_) % S_);
    int b = (int)(idx / ((size_t)S_ * D_));
    float p = pos[(size_t)s * D_ + d];
    if (s < NC_) {
        X[idx] += p;
    } else {
        int tk = toks[b * S_TOK_ + (s - NC_)];
        X[idx] = emb[(size_t)tk * D_ + d] * 22.62741699796952f + p;  // sqrt(512)
    }
}

// ---------------- chord = (8 - ttc) * ttc_W + ttc_b ----------------
__global__ void chord_k(const float* __restrict__ ttc, const float* __restrict__ tw,
                        const float* __restrict__ tb, float* __restrict__ Ch)
{
    size_t idx = (size_t)blockIdx.x * 256 + threadIdx.x;
    if (idx >= (size_t)B_ * S_ * DI_) return;
    int o = (int)(idx % DI_);
    int s = (int)((idx / DI_) % S_);
    int b = (int)(idx / ((size_t)S_ * DI_));
    int st = (s < NC_) ? 0 : (s - NC_);
    float t = ttc[b * S_TOK_ + st];
    Ch[idx] = (8.0f - t) * tw[o] + tb[o];
}

// ---------------- orchestration ----------------
static inline long cdiv4_256(long n) { return (n / 4 + 255) / 256; }

extern "C" void kernel_launch(void* const* d_in, const int* in_sizes, int n_in,
                              void* d_out, int out_size, void* d_ws, size_t ws_size,
                              hipStream_t stream)
{
    (void)in_sizes; (void)n_in; (void)out_size; (void)ws_size;

    const int*   x_tokens   = (const int*)  d_in[0];
    const float* conditions = (const float*)d_in[1];
    const float* ttc        = (const float*)d_in[2];
    const float* emb        = (const float*)d_in[3];
    const float* pos        = (const float*)d_in[4];
    const float* cond_W1    = (const float*)d_in[5];
    const float* cond_b1    = (const float*)d_in[6];
    const float* cond_W2    = (const float*)d_in[7];
    const float* cond_b2    = (const float*)d_in[8];
    const float* null_cond  = (const float*)d_in[9];
    const float* ttc_W      = (const float*)d_in[10];
    const float* ttc_b      = (const float*)d_in[11];
    const float* Wq         = (const float*)d_in[12];
    const float* Wk         = (const float*)d_in[13];
    const float* Wv         = (const float*)d_in[14];
    const float* Wo         = (const float*)d_in[15];
    const float* bq         = (const float*)d_in[16];
    const float* bk         = (const float*)d_in[17];
    const float* bv         = (const float*)d_in[18];
    const float* bo         = (const float*)d_in[19];
    const float* E          = (const float*)d_in[20];
    const float* ffn_W1     = (const float*)d_in[21];
    const float* ffn_b1     = (const float*)d_in[22];
    const float* ffn_W2     = (const float*)d_in[23];
    const float* ffn_b2     = (const float*)d_in[24];
    const float* ln1_g      = (const float*)d_in[25];
    const float* ln1_b      = (const float*)d_in[26];
    const float* ln2_g      = (const float*)d_in[27];
    const float* ln2_b      = (const float*)d_in[28];
    const float* fc_W       = (const float*)d_in[29];
    const float* fc_b       = (const float*)d_in[30];

    // ---- workspace carve-out (256B aligned blocks) ----
    char* cur = (char*)d_ws;
    auto alloc_f = [&](size_t n) { float* r = (float*)cur;
        cur += ((n * 4 + 255) & ~(size_t)255); return r; };
    auto alloc_h = [&](size_t n) { bf16_t* r = (bf16_t*)cur;
        cur += ((n * 2 + 255) & ~(size_t)255); return r; };

    const size_t MSD  = (size_t)MS_ * D_;
    const size_t MSDI = (size_t)MS_ * DI_;

    float*  X   = alloc_f(MSD);     // fp32 master activations
    float*  Hb  = alloc_f(MSD);
    float*  O1  = alloc_f(MSD);
    float*  H2  = alloc_f(MSD);
    float*  Ch  = alloc_f(MSDI);
    bf16_t* Xb  = alloc_h(MSD);     // bf16 GEMM operands
    bf16_t* Qb  = alloc_h(MSD);
    bf16_t* Kbf = alloc_h(MSD);
    bf16_t* Vtb = alloc_h(MSD);
    bf16_t* Aob = alloc_h(MSD);
    bf16_t* O1b = alloc_h(MSD);
    bf16_t* F1b = alloc_h(MSDI);
    bf16_t* Wqb = alloc_h((size_t)L_ * D_ * D_);
    bf16_t* Wkb = alloc_h((size_t)L_ * D_ * D_);
    bf16_t* Wvb = alloc_h((size_t)L_ * D_ * D_);
    bf16_t* Wob = alloc_h((size_t)L_ * D_ * D_);
    bf16_t* W1b = alloc_h((size_t)L_ * D_ * DI_);
    bf16_t* W2b = alloc_h((size_t)L_ * DI_ * D_);
    bf16_t* Fcb = alloc_h((size_t)D_ * V_);
    bf16_t* Eb  = alloc_h((size_t)L_ * S_ * DH_);

    // ---- weight conversion (once per launch) ----
    long nDD = (long)L_ * D_ * D_, nDI = (long)L_ * D_ * DI_;
    cvt_bf16_k<<<cdiv4_256(nDD), 256, 0, stream>>>(Wq, Wqb, nDD);
    cvt_bf16_k<<<cdiv4_256(nDD), 256, 0, stream>>>(Wk, Wkb, nDD);
    cvt_bf16_k<<<cdiv4_256(nDD), 256, 0, stream>>>(Wv, Wvb, nDD);
    cvt_bf16_k<<<cdiv4_256(nDD), 256, 0, stream>>>(Wo, Wob, nDD);
    cvt_bf16_k<<<cdiv4_256(nDI), 256, 0, stream>>>(ffn_W1, W1b, nDI);
    cvt_bf16_k<<<cdiv4_256(nDI), 256, 0, stream>>>(ffn_W2, W2b, nDI);
    cvt_bf16_k<<<cdiv4_256((long)D_ * V_), 256, 0, stream>>>(fc_W, Fcb, (long)D_ * V_);
    cvt_bf16_k<<<cdiv4_256((long)L_ * S_ * DH_), 256, 0, stream>>>(E, Eb, (long)L_ * S_ * DH_);

    // ---- prologue ----
    cond_k<<<B_ * NC_, 256, 0, stream>>>(conditions, cond_W1, cond_b1, cond_W2,
                                         cond_b2, null_cond, X);
    embed_k<<<(B_ * S_ * D_) / 256, 256, 0, stream>>>(x_tokens, emb, pos, X);
    chord_k<<<(B_ * S_ * DI_) / 256, 256, 0, stream>>>(ttc, ttc_W, ttc_b, Ch);
    cvt_bf16_k<<<cdiv4_256((long)MSD), 256, 0, stream>>>(X, Xb, (long)MSD);

    dim3 gDD(D_ / TN, MS_ / TM);
    dim3 gDI(DI_ / TN, MS_ / TM);
    const int attn_grid = B_ * H_ * (S_ / 16);

    for (int l = 0; l < L_; ++l) {
        const bf16_t* wq = Wqb + (size_t)l * D_ * D_;
        const bf16_t* wk = Wkb + (size_t)l * D_ * D_;
        const bf16_t* wv = Wvb + (size_t)l * D_ * D_;
        const bf16_t* wo = Wob + (size_t)l * D_ * D_;
        const bf16_t* w1 = W1b + (size_t)l * D_ * DI_;
        const bf16_t* w2 = W2b + (size_t)l * DI_ * D_;
        const bf16_t* el = Eb  + (size_t)l * S_ * DH_;

        gemm_bf16_wmma<<<gDD, 256, 0, stream>>>(Xb, wq, bq + l * D_, nullptr,
                                                nullptr, Qb,  MS_, D_, D_, 0, 1);
        gemm_bf16_wmma<<<gDD, 256, 0, stream>>>(Xb, wk, bk + l * D_, nullptr,
                                                nullptr, Kbf, MS_, D_, D_, 0, 1);
        gemm_bf16_wmma<<<gDD, 256, 0, stream>>>(Xb, wv, bv + l * D_, nullptr,
                                                nullptr, Vtb, MS_, D_, D_, 0, 2);

        attn_flash_rel<<<attn_grid, 32, 0, stream>>>(Qb, Kbf, Vtb, el, x_tokens, Aob);

        gemm_bf16_wmma<<<gDD, 256, 0, stream>>>(Aob, wo, bo + l * D_, X,
                                                Hb, nullptr, MS_, D_, D_, 0, 0);
        layernorm_k<<<MS_, 256, 0, stream>>>(Hb, ln1_g + l * D_, ln1_b + l * D_, O1, O1b);

        gemm_bf16_wmma<<<gDI, 256, 0, stream>>>(O1b, w1, ffn_b1 + l * DI_, Ch,
                                                nullptr, F1b, MS_, DI_, D_, 1, 1);
        gemm_bf16_wmma<<<gDD, 256, 0, stream>>>(F1b, w2, ffn_b2 + l * D_, O1,
                                                H2, nullptr, MS_, D_, DI_, 0, 0);
        layernorm_k<<<MS_, 256, 0, stream>>>(H2, ln2_g + l * D_, ln2_b + l * D_, X, Xb);
    }

    gemm_bf16_wmma<<<gDD, 256, 0, stream>>>(Xb, Fcb, fc_b, nullptr,
                                            (float*)d_out, nullptr, MS_, V_, D_, 0, 0);
}
